// GraphFeatureExtractor_25340307046637
// MI455X (gfx1250) — matrix operand order repaired and checked by reading, not compiled, not run on hardware
//
#include <hip/hip_runtime.h>

typedef __attribute__((ext_vector_type(16))) _Float16 v16h;
typedef __attribute__((ext_vector_type(8)))  _Float16 v8h;
typedef __attribute__((ext_vector_type(8)))  float    v8f;

#define DIN   128
#define RELS  4
#define KTOT  640   // DIN * (RELS + 1)

// ---------------- zero fill ----------------
__global__ void k_zero(float* __restrict__ p, long n) {
  long i = (long)blockIdx.x * blockDim.x + threadIdx.x;
  long s = (long)gridDim.x * blockDim.x;
  for (; i < n; i += s) p[i] = 0.0f;
}

// ---------------- edge scatter: wave per edge ----------------
// agg[(dst*RELS + r)*DIN + k] += x[src][k];  deg[dst*RELS + r] += 1
__global__ void k_scatter(const float* __restrict__ x,
                          const int* __restrict__ ei,   // [2*E], src then dst
                          const int* __restrict__ et,   // [E]
                          float* __restrict__ agg,
                          float* __restrict__ deg,
                          int E) {
  int lane  = threadIdx.x & 31;
  long tid  = (long)blockIdx.x * blockDim.x + threadIdx.x;
  int wave  = (int)(tid >> 5);
  int nwave = (int)(((long)gridDim.x * blockDim.x) >> 5);
  for (int e = wave; e < E; e += nwave) {
    int s = ei[e];
    int d = ei[E + e];
    int r = et[e];
    const float4 v = *(const float4*)(x + (size_t)s * DIN + lane * 4);
    float* dp = agg + ((size_t)d * RELS + r) * DIN + lane * 4;
    unsafeAtomicAdd(dp + 0, v.x);
    unsafeAtomicAdd(dp + 1, v.y);
    unsafeAtomicAdd(dp + 2, v.z);
    unsafeAtomicAdd(dp + 3, v.w);
    if (lane == 0) unsafeAtomicAdd(deg + (size_t)d * RELS + r, 1.0f);
  }
}

// ---------------- build fused A matrix (N x 640, f16) ----------------
// Ah[i][0:128] = x[i];  Ah[i][128 + r*128 + k] = agg[i][r][k] / max(deg,1)
__global__ void k_build_a(const float* __restrict__ x,
                          const float* __restrict__ agg,
                          const float* __restrict__ deg,
                          _Float16* __restrict__ Ah, long N) {
  long i = (long)blockIdx.x * blockDim.x + threadIdx.x;
  long total = N * KTOT;
  long s = (long)gridDim.x * blockDim.x;
  for (; i < total; i += s) {
    long node = i / KTOT;
    int  j    = (int)(i % KTOT);
    float v;
    if (j < DIN) {
      v = x[node * DIN + j];
    } else {
      int r = (j - DIN) >> 7;
      int k = (j - DIN) & 127;
      float dg = deg[node * RELS + r];
      v = agg[(node * RELS + r) * DIN + k] / fmaxf(dg, 1.0f);
    }
    Ah[i] = (_Float16)v;
  }
}

// ---------------- build transposed stacked weights (NOUT x 640, f16) ----------------
// Bt[n][k] = (k < 128 ? W_root[k][n] : W_rel[(k-128)/128][(k-128)%128][n])
__global__ void k_build_bt(const float* __restrict__ Wroot,
                           const float* __restrict__ Wrel,
                           _Float16* __restrict__ Bt, int NOUT) {
  int idx = blockIdx.x * blockDim.x + threadIdx.x;
  int total = NOUT * KTOT;
  if (idx >= total) return;
  int n = idx / KTOT;
  int k = idx % KTOT;
  float v;
  if (k < DIN) {
    v = Wroot[(size_t)k * NOUT + n];
  } else {
    int r  = (k - DIN) >> 7;
    int kk = (k - DIN) & 127;
    v = Wrel[((size_t)r * DIN + kk) * NOUT + n];
  }
  Bt[idx] = (_Float16)v;
}

// ---------------- WMMA GEMM:  out = relu?(Ah @ B + bias) ----------------
// Block: 128 rows x (32*CT) cols.  8 waves in 4x2 grid; wave = 32 rows x (16*CT) cols.
template <int CT, bool RELU>
__global__ __launch_bounds__(256) void k_gemm(
    const _Float16* __restrict__ A,    // M x KTOT, row major
    const _Float16* __restrict__ Bt,   // NOUT x KTOT (i.e. B column-major)
    const float* __restrict__ bias,    // NOUT
    float* __restrict__ out,           // M x NOUT
    int M) {
  const int NOUT = 32 * CT;
  int wid  = threadIdx.x >> 5;
  int lane = threadIdx.x & 31;
  int lrow = lane & 15;
  int hs   = lane >> 4;          // half-wave select
  int wrow = wid >> 1;           // 0..3
  int wcol = wid & 1;            // 0..1

  int m_base = blockIdx.x * 128 + wrow * 32;
  int c_base = wcol * (16 * CT);

  // clamp A row indices so every lane issues identical control flow (EXEC all 1s)
  int r0 = m_base + lrow;        if (r0 >= M) r0 = M - 1;
  int r1 = m_base + 16 + lrow;   if (r1 >= M) r1 = M - 1;
  const _Float16* arow0 = A + (size_t)r0 * KTOT;
  const _Float16* arow1 = A + (size_t)r1 * KTOT;

  v8f acc[2][CT];
#pragma unroll
  for (int i = 0; i < 2; ++i)
#pragma unroll
    for (int j = 0; j < CT; ++j) acc[i][j] = v8f{};

  for (int k0 = 0; k0 < KTOT; k0 += 32) {
    // A fragments (contiguous K order: lanes<16 -> K 0..15, lanes>=16 -> K 16..31)
    union { v16h v; v8h h[2]; } ua0, ua1;
    {
      const _Float16* p0 = arow0 + k0 + hs * 16;
      ua0.h[0] = *(const v8h*)(p0);
      ua0.h[1] = *(const v8h*)(p0 + 8);
      const _Float16* p1 = arow1 + k0 + hs * 16;
      ua1.h[0] = *(const v8h*)(p1);
      ua1.h[1] = *(const v8h*)(p1 + 8);
    }
#pragma unroll
    for (int ct = 0; ct < CT; ++ct) {
      int c = c_base + ct * 16 + lrow;
      // B fragment (contiguous K order, as A): lane = column, lanes<16 hold K 0..15, lanes>=16 hold K 16..31
      v16h bfrag = *(const v16h*)(Bt + (size_t)c * KTOT + k0 + hs * 16);
      acc[0][ct] = __builtin_amdgcn_wmma_f32_16x16x32_f16(
          false, ua0.v, false, bfrag, (short)0, acc[0][ct], false, false);
      acc[1][ct] = __builtin_amdgcn_wmma_f32_16x16x32_f16(
          false, ua1.v, false, bfrag, (short)0, acc[1][ct], false, false);
    }
  }

  // store: C/D layout -> VGPR v holds row (hs?8:0)+v, col = lane%16
#pragma unroll
  for (int rt = 0; rt < 2; ++rt) {
#pragma unroll
    for (int ct = 0; ct < CT; ++ct) {
      int c = c_base + ct * 16 + lrow;
      float bv = bias[c];
#pragma unroll
      for (int v = 0; v < 8; ++v) {
        int row = m_base + rt * 16 + hs * 8 + v;
        if (row < M) {
          float val = acc[rt][ct][v] + bv;
          if (RELU) val = fmaxf(val, 0.0f);
          out[(size_t)row * NOUT + c] = val;
        }
      }
    }
  }
}

extern "C" void kernel_launch(void* const* d_in, const int* in_sizes, int n_in,
                              void* d_out, int out_size, void* d_ws, size_t ws_size,
                              hipStream_t stream) {
  const float* x     = (const float*)d_in[0];
  const int*   ei    = (const int*)d_in[1];
  const int*   et    = (const int*)d_in[2];
  const float* Wr1   = (const float*)d_in[3];
  const float* Wrel1 = (const float*)d_in[4];
  const float* b1    = (const float*)d_in[5];
  const float* Wr2   = (const float*)d_in[6];
  const float* Wrel2 = (const float*)d_in[7];
  const float* b2    = (const float*)d_in[8];
  float* out = (float*)d_out;

  const long N = in_sizes[0] / DIN;     // 50000
  const int  E = in_sizes[1] / 2;       // 600000
  (void)n_in; (void)out_size; (void)ws_size;

  // workspace carve-up (256B aligned)
  char* ws = (char*)d_ws;
  size_t off = 0;
  auto carve = [&](size_t bytes) -> void* {
    void* p = ws + off;
    off += (bytes + 255) & ~(size_t)255;
    return p;
  };
  float*    agg = (float*)   carve((size_t)N * RELS * DIN * sizeof(float)); // 102.4 MB
  float*    deg = (float*)   carve((size_t)N * RELS * sizeof(float));
  _Float16* Ah  = (_Float16*)carve((size_t)N * KTOT * sizeof(_Float16));    // 64 MB
  float*    h   = (float*)   carve((size_t)N * DIN * sizeof(float));        // 25.6 MB
  _Float16* Bt1 = (_Float16*)carve((size_t)DIN * KTOT * sizeof(_Float16));
  _Float16* Bt2 = (_Float16*)carve((size_t)96  * KTOT * sizeof(_Float16));

  const long aggN = N * RELS * DIN;
  const long degN = N * RELS;
  const int  ZB = 2048, T = 256;
  const int  scatterBlocks = (E + 7) / 8;             // 8 waves / block
  const int  buildBlocks   = 8192;
  const int  gemmBlocks    = (int)((N + 127) / 128);

  // weight prep (tiny)
  k_build_bt<<<(DIN * KTOT + T - 1) / T, T, 0, stream>>>(Wr1, Wrel1, Bt1, DIN);
  k_build_bt<<<(96  * KTOT + T - 1) / T, T, 0, stream>>>(Wr2, Wrel2, Bt2, 96);

  // ---- layer 1 ----
  k_zero<<<ZB, T, 0, stream>>>(agg, aggN);
  k_zero<<<ZB, T, 0, stream>>>(deg, degN);
  k_scatter<<<scatterBlocks, T, 0, stream>>>(x, ei, et, agg, deg, E);
  k_build_a<<<buildBlocks, T, 0, stream>>>(x, agg, deg, Ah, N);
  k_gemm<4, true><<<gemmBlocks, T, 0, stream>>>(Ah, Bt1, b1, h, (int)N);

  // ---- layer 2 ----
  k_zero<<<ZB, T, 0, stream>>>(agg, aggN);
  k_zero<<<ZB, T, 0, stream>>>(deg, degN);
  k_scatter<<<scatterBlocks, T, 0, stream>>>(h, ei, et, agg, deg, E);
  k_build_a<<<buildBlocks, T, 0, stream>>>(h, agg, deg, Ah, N);
  k_gemm<3, false><<<gemmBlocks, T, 0, stream>>>(Ah, Bt2, b2, out, (int)N);
}
